// TravelTimeCrossCorrelation_49778670960692
// MI455X (gfx1250) — matrix operand order repaired and checked by reading, not compiled, not run on hardware
//
#include <hip/hip_runtime.h>
#include <hip/hip_bf16.h>

typedef __attribute__((ext_vector_type(16))) _Float16 v16h;
typedef __attribute__((ext_vector_type(8)))  float    v8f;

#define NB 4
#define NT 2048
#define NR 64
#define NC 16
#define NTR (NB * NR * NC)        // 4096 traces
#define LL (2 * NT - 1)           // 4095 lags
#define CENTER (NT - 1)           // 2047
#define SCALE 100.0f
#define DTC 0.001f

// padded LDS staging sizes (halves)
#define XOFF 64
#define XLEN (NT + 128)           // 2176
#define YOFF 256
#define YLEN (NT + 512)           // 2560

// One workgroup per trace: 256 threads = 8 waves; 16 lag-tiles of 256 lags; 2 tiles/wave.
__global__ __launch_bounds__(256)
void ttcc_corr_argmax(const float* __restrict__ x,
                      const float* __restrict__ y,
                      const float* __restrict__ gu,
                      float* __restrict__ ws) {
    __shared__ _Float16 sx_hi[XLEN], sx_lo[XLEN];
    __shared__ _Float16 sy_hi[YLEN], sy_lo[YLEN];
    __shared__ float    rmax[256];
    __shared__ int      ridx[256];
    __shared__ int      s_xnz, s_ynz;

    const int tid   = threadIdx.x;
    const int trace = blockIdx.x;                 // [0, 4096)
    const int b  = trace >> 10;                   // nr*nc = 1024
    const int rc = trace & 1023;
    const int r  = rc >> 4;
    const int c  = rc & 15;

    // ---- zero padded staging arrays ----
    for (int i = tid; i < XLEN; i += 256) { sx_hi[i] = (_Float16)0; sx_lo[i] = (_Float16)0; }
    for (int i = tid; i < YLEN; i += 256) { sy_hi[i] = (_Float16)0; sy_lo[i] = (_Float16)0; }
    if (tid == 0) { s_xnz = 0; s_ynz = 0; }
    __syncthreads();

    // ---- stage trace into LDS as hi/lo f16 split (x[b, t, r, c], stride over t = 1024) ----
    const size_t base = (size_t)b * (size_t)(NT * NR * NC) + (size_t)(r * NC + c);
    const float* xg = x + base;
    const float* yg = y + base;
    for (int t = tid; t < NT; t += 256) {
        float vx = xg[(size_t)t * (NR * NC)];
        _Float16 hx = (_Float16)vx;
        sx_hi[XOFF + t] = hx;
        sx_lo[XOFF + t] = (_Float16)(vx - (float)hx);
        if (vx != 0.0f) s_xnz = 1;                // benign race: writes same value
        float vy = yg[(size_t)t * (NR * NC)];
        _Float16 hy = (_Float16)vy;
        sy_hi[YOFF + t] = hy;
        sy_lo[YOFF + t] = (_Float16)(vy - (float)hy);
        if (vy != 0.0f) s_ynz = 1;
    }
    __syncthreads();

    const int wave = __builtin_amdgcn_readfirstlane(tid >> 5);  // 0..7, force SGPR
    const int lane = tid & 31;
    const int arow = lane & 15;                   // A: row M = lane & 15
    const int akb  = (lane >> 4) << 3;            // A: K half-base 0 or 8
    const int bcol = lane & 15;                   // B: column N = lane & 15
    const int bkb  = (lane >> 4) << 4;            // B: K base 0 or 16
    const int crow = (lane >> 4) << 3;            // C: M = v + (lane>=16 ? 8 : 0)

    const float* ug = gu + (size_t)trace * (size_t)LL;
    __builtin_prefetch(ug, 0, 1);                 // global_prefetch_b8

    float bmax = -3.4e38f;
    int   bidx = 0x7fffffff;

    for (int ti = 0; ti < 2; ++ti) {
        const int tile  = wave * 2 + ti;          // 0..15, uniform per wave
        const int jbase = tile << 8;              // 256 lags per tile

        // K support where both operands can be nonzero, aligned to 32 (scalar/uniform)
        int klo = 2032 - jbase; if (klo < 0) klo = 0; klo &= ~31;
        int khi = 4096 - jbase; if (khi > 2288) khi = 2288; khi = (khi + 31) & ~31;
        klo = __builtin_amdgcn_readfirstlane(klo);
        khi = __builtin_amdgcn_readfirstlane(khi);

        // independent accumulators -> no C-operand chain between the 3 WMMAs
        v8f acc_hh = {}, acc_hl = {}, acc_lh = {};
        const int abase0 = XOFF + arow + jbase - CENTER;  // + k0 + K
        const int bbase0 = YOFF - (bcol << 4);            // + k0 + K

        for (int k0 = klo; k0 < khi; k0 += 32) {
            v16h ah, al, bh, bl;
            const int abase = abase0 + k0;
#pragma unroll
            for (int e = 0; e < 16; ++e) {
                // A 16x32 f16 layout: element e -> K = ((e&8)<<1) + akb + (e&7)
                int K = ((e & 8) << 1) + akb + (e & 7);
                ah[e] = sx_hi[abase + K];
                al[e] = sx_lo[abase + K];
            }
            const int bbase = bbase0 + k0;
#pragma unroll
            for (int e = 0; e < 16; ++e) {
                // B 32x16 f16 layout: element e -> K = ((e&8)<<1) + (bkb>>1) + (e&7)
                int K = ((e & 8) << 1) + (bkb >> 1) + (e & 7);
                bh[e] = sy_hi[bbase + K];
                bl[e] = sy_lo[bbase + K];
            }
            // hi*hi + hi*lo + lo*hi  (near-fp32 accuracy, drops lo*lo ~2^-22)
            acc_hh = __builtin_amdgcn_wmma_f32_16x16x32_f16(false, ah, false, bh,
                                                            (short)0, acc_hh, false, false);
            acc_hl = __builtin_amdgcn_wmma_f32_16x16x32_f16(false, ah, false, bl,
                                                            (short)0, acc_hl, false, false);
            acc_lh = __builtin_amdgcn_wmma_f32_16x16x32_f16(false, al, false, bh,
                                                            (short)0, acc_lh, false, false);
        }

        // epilogue: z = cc*scale + gumbel, running argmax (tie -> lowest lag)
#pragma unroll
        for (int v = 0; v < 8; ++v) {
            int lag = jbase + (v + crow) + (bcol << 4);   // jbase + M + 16*N
            if (lag < LL) {
                float cc = acc_hh[v] + (acc_hl[v] + acc_lh[v]);
                float u = ug[lag];
                u = fminf(fmaxf(u, 1e-7f), 1.0f - 1e-7f);
                float g = -__logf(-__logf(u));
                float z = cc * SCALE + g;
                if (z > bmax || (z == bmax && lag < bidx)) { bmax = z; bidx = lag; }
            }
        }
    }

    // ---- workgroup argmax reduction ----
    rmax[tid] = bmax;
    ridx[tid] = bidx;
    __syncthreads();
    for (int s = 128; s > 0; s >>= 1) {
        if (tid < s) {
            float ov = rmax[tid + s]; int oi = ridx[tid + s];
            if (ov > rmax[tid] || (ov == rmax[tid] && oi < ridx[tid])) {
                rmax[tid] = ov; ridx[tid] = oi;
            }
        }
        __syncthreads();
    }
    if (tid == 0) {
        float contrib = (s_xnz && s_ynz) ? (float)(ridx[0] - CENTER) * DTC : 0.0f;
        ws[trace] = contrib;
    }
}

// Deterministic single-block sum of per-trace contributions.
__global__ __launch_bounds__(256)
void ttcc_reduce(const float* __restrict__ ws, float* __restrict__ out) {
    __shared__ float red[256];
    const int tid = threadIdx.x;
    float s = 0.0f;
    for (int i = tid; i < NTR; i += 256) s += ws[i];
    red[tid] = s;
    __syncthreads();
    for (int st = 128; st > 0; st >>= 1) {
        if (tid < st) red[tid] += red[tid + st];
        __syncthreads();
    }
    if (tid == 0) out[0] = red[0];
}

extern "C" void kernel_launch(void* const* d_in, const int* in_sizes, int n_in,
                              void* d_out, int out_size, void* d_ws, size_t ws_size,
                              hipStream_t stream) {
    const float* x  = (const float*)d_in[0];
    const float* y  = (const float*)d_in[1];
    const float* gu = (const float*)d_in[2];
    float* out = (float*)d_out;
    float* ws  = (float*)d_ws;   // 4096 floats = 16 KB of scratch

    ttcc_corr_argmax<<<NTR, 256, 0, stream>>>(x, y, gu, ws);
    ttcc_reduce<<<1, 256, 0, stream>>>(ws, out);
}
